// SWTB_34866544509457
// MI455X (gfx1250) — matrix operand order repaired and checked by reading, not compiled, not run on hardware
//
#include <hip/hip_runtime.h>
#include <hip/hip_bf16.h>

// ---------------------------------------------------------------------------
// Swin Transformer block, MI455X (gfx1250): all GEMMs via v_wmma_f32_16x16x32_bf16
// B=16, C=256, H=W=64, WS=8 (N=64 tokens/window), NH=8, hd=32
// GEMMs: 128x64 block tile (8 waves), B-operand double-buffered through LDS.
// ---------------------------------------------------------------------------

typedef __attribute__((ext_vector_type(16))) __bf16 bf16x16;
typedef __attribute__((ext_vector_type(8)))  __bf16 bf16x8;
typedef __attribute__((ext_vector_type(8)))  float  f32x8;

__device__ __forceinline__ f32x8 wmma_bf16(bf16x16 a, bf16x16 b, f32x8 c) {
  return __builtin_amdgcn_wmma_f32_16x16x32_bf16(false, a, false, b, (short)0, c,
                                                 false, false);
}

// A fragment (16x32, MxK, 16-bit): lane = M (0..15, both halves), halves:
//   lanes 0-15 : a[0..7]=K 0..7,  a[8..15]=K16..23
//   lanes 16-31: a[0..7]=K 8..15, a[8..15]=K24..31
__device__ __forceinline__ bf16x16 load_a_frag(const __bf16* A, int lda, int lane) {
  int m  = lane & 15;
  int kb = (lane & 16) >> 1;          // 0 or 8
  const __bf16* p = A + (size_t)m * lda + kb;
  bf16x8 lo = *(const bf16x8*)(p);
  bf16x8 hi = *(const bf16x8*)(p + 16);
  return __builtin_shufflevector(lo, hi, 0,1,2,3,4,5,6,7,8,9,10,11,12,13,14,15);
}

// B fragment (32x16, KxN, 16-bit), from row-major weights W[n][k] (B[k][n]=W[n][k]):
//   lane = N (0..15), lanes 0-15: b[0..7]=K 0..7, b[8..15]=K16..23; lanes 16-31: b[0..7]=K 8..15, b[8..15]=K24..31
__device__ __forceinline__ bf16x16 load_b_frag(const __bf16* W, int ldw, int lane) {
  int n  = lane & 15;
  int kb = (lane & 16) >> 1;           // 0 or 8
  const __bf16* p = W + (size_t)n * ldw + kb;
  bf16x8 lo = *(const bf16x8*)(p);
  bf16x8 hi = *(const bf16x8*)(p + 16);
  return __builtin_shufflevector(lo, hi, 0,1,2,3,4,5,6,7,8,9,10,11,12,13,14,15);
}

// ---------------------------------------------------------------------------
// Block GEMM core: 128x64 output tile per block (8 waves, each 16x64 strip).
// A[m][k] row-major bf16 (per-wave global loads); W[n][k] row-major bf16,
// 64x32 B slab per K-step staged to LDS, double-buffered.
// ---------------------------------------------------------------------------
template <int KSTEPS>
__device__ __forceinline__ void gemm_128x64(const __bf16* __restrict__ A, int lda,
                                            const __bf16* __restrict__ W, int ldw,
                                            __bf16 (*lds)[2048], f32x8 acc[4]) {
  int tid = threadIdx.x, warp = tid >> 5, lane = tid & 31;
  int srow = tid >> 2, scol = (tid & 3) * 8;            // 4 threads per B row
  const __bf16* Ap = A + (size_t)(warp * 16) * lda;
  // prologue: stage K-step 0
  {
    bf16x8 v = *(const bf16x8*)(W + (size_t)srow * ldw + scol);
    *(bf16x8*)(&lds[0][srow * 32 + scol]) = v;
  }
  __syncthreads();
#pragma unroll 4
  for (int ks = 0; ks < KSTEPS; ++ks) {
    int cur = ks & 1;
    if (ks + 1 < KSTEPS) {                              // stage next slab
      bf16x8 v = *(const bf16x8*)(W + (size_t)srow * ldw + (ks + 1) * 32 + scol);
      *(bf16x8*)(&lds[cur ^ 1][srow * 32 + scol]) = v;
    }
    bf16x16 a = load_a_frag(Ap + ks * 32, lda, lane);
    bf16x16 bfr[4];
#pragma unroll
    for (int nt = 0; nt < 4; ++nt)
      bfr[nt] = load_b_frag(&lds[cur][nt * 16 * 32], 32, lane);
#pragma unroll
    for (int nt = 0; nt < 4; ++nt)
      acc[nt] = wmma_bf16(a, bfr[nt], acc[nt]);
    __syncthreads();
  }
}

// ---------------------------------------------------------------------------
// K0: weights fp32 -> bf16, expand rel-pos bias to fp32 [NH][64][64]
// ---------------------------------------------------------------------------
__global__ void swin_prep(const float* qkv_w, const float* proj_w,
                          const float* fc1_w, const float* fc2_w,
                          const float* table, const int* rel_index,
                          __bf16* wq, __bf16* wp, __bf16* w1, __bf16* w2,
                          float* bias_out) {
  int i = blockIdx.x * blockDim.x + threadIdx.x;       // up to 262144
  if (i < 196608) wq[i] = (__bf16)qkv_w[i];
  if (i < 65536)  wp[i] = (__bf16)proj_w[i];
  if (i < 262144) w1[i] = (__bf16)fc1_w[i];
  if (i < 262144) w2[i] = (__bf16)fc2_w[i];
  if (i < 32768) {                                     // 8 heads * 4096 (i,j)
    int nh = i & 7, ij = i >> 3;
    bias_out[nh * 4096 + ij] = table[rel_index[ij] * 8 + nh];
  }
}

// ---------------------------------------------------------------------------
// K1: layernorm1 + window partition; one wave per token, write bf16
// ---------------------------------------------------------------------------
__global__ void swin_ln1(const float* __restrict__ x, const float* __restrict__ g,
                         const float* __restrict__ bb, __bf16* __restrict__ y_win) {
  int warp = threadIdx.x >> 5, lane = threadIdx.x & 31;
  int t  = blockIdx.x * 8 + warp;                      // token 0..65535
  int b  = t >> 12, hw = t & 4095, h = hw >> 6, w = hw & 63;
  const float* xp = x + (size_t)b * 256 * 4096 + hw;   // + c*4096
  float vals[8], s = 0.f, sq = 0.f;
#pragma unroll
  for (int i = 0; i < 8; ++i) {
    int c = lane + i * 32;
    float v = xp[(size_t)c * 4096];
    vals[i] = v; s += v; sq += v * v;
  }
#pragma unroll
  for (int m = 1; m < 32; m <<= 1) { s += __shfl_xor(s, m, 32); sq += __shfl_xor(sq, m, 32); }
  float mean = s * (1.f / 256.f);
  float var  = sq * (1.f / 256.f) - mean * mean;
  float rstd = rsqrtf(var + 1e-5f);
  int wi = (b << 6) + ((h >> 3) << 3) + (w >> 3);
  int r  = ((h & 7) << 3) + (w & 7);
  __bf16* yp = y_win + ((size_t)wi * 64 + r) * 256;
#pragma unroll
  for (int i = 0; i < 8; ++i) {
    int c = lane + i * 32;
    yp[c] = (__bf16)((vals[i] - mean) * rstd * g[c] + bb[c]);
  }
}

// ---------------------------------------------------------------------------
// K2: QKV GEMM  [65536 x 256] @ [768 x 256]^T + bias -> bf16 [65536 x 768]
// ---------------------------------------------------------------------------
__global__ void swin_qkv(const __bf16* __restrict__ A, const __bf16* __restrict__ W,
                         const float* __restrict__ bias, __bf16* __restrict__ out) {
  __shared__ __bf16 lds[2][2048];
  int blk = blockIdx.x;
  int mt = blk / 12, nb = blk % 12;
  int m0 = mt * 128, n0 = nb * 64;
  f32x8 acc[4] = {};
  gemm_128x64<8>(A + (size_t)m0 * 256, 256, W + (size_t)n0 * 256, 256, lds, acc);
  int warp = threadIdx.x >> 5, lane = threadIdx.x & 31;
  int n = lane & 15, mh = (lane & 16) >> 1;
#pragma unroll
  for (int nt = 0; nt < 4; ++nt)
#pragma unroll
    for (int v = 0; v < 8; ++v) {
      int col = n0 + nt * 16 + n, row = m0 + warp * 16 + mh + v;
      out[(size_t)row * 768 + col] = (__bf16)(acc[nt][v] + bias[col]);
    }
}

// ---------------------------------------------------------------------------
// K3: windowed attention; one block (4 waves) per (window, head)
// ---------------------------------------------------------------------------
__global__ void swin_attn(const __bf16* __restrict__ qkv, const float* __restrict__ bias,
                          __bf16* __restrict__ attn_out) {
  __shared__ __bf16 Vt[32 * 64];        // Vt[d][j] = V[j][d]
  __shared__ __bf16 P[4 * 16 * 64];     // softmaxed scores, row-major per wave
  int tid = threadIdx.x, warp = tid >> 5, lane = tid & 31;
  int wh = blockIdx.x, wi = wh >> 3, nh = wh & 7;
  const __bf16* qb = qkv + (size_t)wi * 64 * 768 + nh * 32;   // q[row][d], ld 768
  const __bf16* kb = qb + 256;
  const __bf16* vb = qb + 512;
  for (int e = tid; e < 2048; e += 128) {               // transpose V into LDS
    int j = e >> 5, d = e & 31;
    Vt[d * 64 + j] = vb[(size_t)j * 768 + d];
  }
  __syncthreads();

  // scores: 16x64 strip per wave; hd=32 -> single K-step
  f32x8 acc[4];
  bf16x16 aq = load_a_frag(qb + (size_t)warp * 16 * 768, 768, lane);
  f32x8 zero = {};
#pragma unroll
  for (int nt = 0; nt < 4; ++nt) {
    bf16x16 bk = load_b_frag(kb + (size_t)nt * 16 * 768, 768, lane);
    acc[nt] = wmma_bf16(aq, bk, zero);
  }
  const float scale = 0.1767766952966369f;              // 1/sqrt(32)
  int n = lane & 15, mh = (lane & 16) >> 1;
#pragma unroll
  for (int v = 0; v < 8; ++v) {
    int mrow = warp * 16 + mh + v;
    float mx = -1e30f;
#pragma unroll
    for (int nt = 0; nt < 4; ++nt) {
      float s = acc[nt][v] * scale + bias[((size_t)nh * 64 + mrow) * 64 + nt * 16 + n];
      acc[nt][v] = s;
      mx = fmaxf(mx, s);
    }
#pragma unroll
    for (int msk = 1; msk < 16; msk <<= 1) mx = fmaxf(mx, __shfl_xor(mx, msk, 32));
    float sm = 0.f;
#pragma unroll
    for (int nt = 0; nt < 4; ++nt) {
      float e = __expf(acc[nt][v] - mx);
      acc[nt][v] = e; sm += e;
    }
#pragma unroll
    for (int msk = 1; msk < 16; msk <<= 1) sm += __shfl_xor(sm, msk, 32);
    float inv = 1.f / sm;
#pragma unroll
    for (int nt = 0; nt < 4; ++nt)
      P[(warp * 16 + mh + v) * 64 + nt * 16 + n] = (__bf16)(acc[nt][v] * inv);
  }
  __syncthreads();

  // out = P @ V : 16x32 per wave, K=64 (2 steps), 2 N-tiles
  f32x8 oacc[2] = {};
#pragma unroll
  for (int kk = 0; kk < 2; ++kk) {
    bf16x16 ap = load_a_frag(&P[warp * 16 * 64 + kk * 32], 64, lane);
#pragma unroll
    for (int n2 = 0; n2 < 2; ++n2) {
      bf16x16 bv = load_b_frag(&Vt[(size_t)n2 * 16 * 64 + kk * 32], 64, lane);
      oacc[n2] = wmma_bf16(ap, bv, oacc[n2]);
    }
  }
  __bf16* op = attn_out + (size_t)wi * 64 * 256 + nh * 32;
#pragma unroll
  for (int n2 = 0; n2 < 2; ++n2)
#pragma unroll
    for (int v = 0; v < 8; ++v)
      op[(size_t)(warp * 16 + mh + v) * 256 + n2 * 16 + n] = (__bf16)(oacc[n2][v]);
}

// ---------------------------------------------------------------------------
// K4: proj GEMM + bias + window-reverse + residual(x) -> fp32 xs2 [65536 x 256]
// ---------------------------------------------------------------------------
__global__ void swin_proj(const __bf16* __restrict__ A, const __bf16* __restrict__ W,
                          const float* __restrict__ bias, const float* __restrict__ x,
                          float* __restrict__ xs2) {
  __shared__ __bf16 lds[2][2048];
  int blk = blockIdx.x;
  int mt = blk >> 2, nb = blk & 3;
  int m0 = mt * 128, n0 = nb * 64;
  f32x8 acc[4] = {};
  gemm_128x64<8>(A + (size_t)m0 * 256, 256, W + (size_t)n0 * 256, 256, lds, acc);
  int warp = threadIdx.x >> 5, lane = threadIdx.x & 31;
  int n = lane & 15, mh = (lane & 16) >> 1;
#pragma unroll
  for (int nt = 0; nt < 4; ++nt)
#pragma unroll
    for (int v = 0; v < 8; ++v) {
      int c  = n0 + nt * 16 + n;
      int rg = m0 + warp * 16 + mh + v;                 // window-order row
      int wi = rg >> 6, r = rg & 63;
      int b  = wi >> 6, wbh = (wi >> 3) & 7, wbw = wi & 7;
      int h  = wbh * 8 + (r >> 3), w = wbw * 8 + (r & 7);
      int t  = (b << 12) + (h << 6) + w;
      float sc = x[((size_t)(b * 256 + c)) * 4096 + (h << 6) + w];
      xs2[(size_t)t * 256 + c] = acc[nt][v] + bias[c] + sc;
    }
}

// ---------------------------------------------------------------------------
// K5: layernorm2 (token-major), fp32 -> bf16
// ---------------------------------------------------------------------------
__global__ void swin_ln2(const float* __restrict__ xs2, const float* __restrict__ g,
                         const float* __restrict__ bb, __bf16* __restrict__ out) {
  int warp = threadIdx.x >> 5, lane = threadIdx.x & 31;
  int t = blockIdx.x * 8 + warp;
  const float* xp = xs2 + (size_t)t * 256;
  float vals[8], s = 0.f, sq = 0.f;
#pragma unroll
  for (int i = 0; i < 8; ++i) {
    float v = xp[lane + i * 32];
    vals[i] = v; s += v; sq += v * v;
  }
#pragma unroll
  for (int m = 1; m < 32; m <<= 1) { s += __shfl_xor(s, m, 32); sq += __shfl_xor(sq, m, 32); }
  float mean = s * (1.f / 256.f);
  float var  = sq * (1.f / 256.f) - mean * mean;
  float rstd = rsqrtf(var + 1e-5f);
  __bf16* yp = out + (size_t)t * 256;
#pragma unroll
  for (int i = 0; i < 8; ++i) {
    int c = lane + i * 32;
    yp[c] = (__bf16)((vals[i] - mean) * rstd * g[c] + bb[c]);
  }
}

// ---------------------------------------------------------------------------
// K6: FC1 GEMM [65536x256]@[1024x256]^T + bias + exact GELU -> bf16
// ---------------------------------------------------------------------------
__global__ void swin_fc1(const __bf16* __restrict__ A, const __bf16* __restrict__ W,
                         const float* __restrict__ bias, __bf16* __restrict__ out) {
  __shared__ __bf16 lds[2][2048];
  int blk = blockIdx.x;
  int mt = blk >> 4, nb = blk & 15;
  int m0 = mt * 128, n0 = nb * 64;
  f32x8 acc[4] = {};
  gemm_128x64<8>(A + (size_t)m0 * 256, 256, W + (size_t)n0 * 256, 256, lds, acc);
  int warp = threadIdx.x >> 5, lane = threadIdx.x & 31;
  int n = lane & 15, mh = (lane & 16) >> 1;
#pragma unroll
  for (int nt = 0; nt < 4; ++nt)
#pragma unroll
    for (int v = 0; v < 8; ++v) {
      int col = n0 + nt * 16 + n, row = m0 + warp * 16 + mh + v;
      float z = acc[nt][v] + bias[col];
      float gl = 0.5f * z * (1.0f + erff(z * 0.7071067811865476f));
      out[(size_t)row * 1024 + col] = (__bf16)gl;
    }
}

// ---------------------------------------------------------------------------
// K7: FC2 GEMM [65536x1024]@[256x1024]^T + bias + residual + NCHW permute out
// ---------------------------------------------------------------------------
__global__ void swin_fc2(const __bf16* __restrict__ A, const __bf16* __restrict__ W,
                         const float* __restrict__ bias, const float* __restrict__ xs2,
                         float* __restrict__ out) {
  __shared__ __bf16 lds[2][2048];
  int blk = blockIdx.x;
  int mt = blk >> 2, nb = blk & 3;
  int m0 = mt * 128, n0 = nb * 64;
  f32x8 acc[4] = {};
  gemm_128x64<32>(A + (size_t)m0 * 1024, 1024, W + (size_t)n0 * 1024, 1024, lds, acc);
  int warp = threadIdx.x >> 5, lane = threadIdx.x & 31;
  int n = lane & 15, mh = (lane & 16) >> 1;
#pragma unroll
  for (int nt = 0; nt < 4; ++nt)
#pragma unroll
    for (int v = 0; v < 8; ++v) {
      int c = n0 + nt * 16 + n;
      int t = m0 + warp * 16 + mh + v;
      int b = t >> 12, hw = t & 4095;
      float val = acc[nt][v] + bias[c] + xs2[(size_t)t * 256 + c];
      out[((size_t)(b * 256 + c)) * 4096 + hw] = val;
    }
}

// ---------------------------------------------------------------------------
extern "C" void kernel_launch(void* const* d_in, const int* in_sizes, int n_in,
                              void* d_out, int out_size, void* d_ws, size_t ws_size,
                              hipStream_t stream) {
  const float* x        = (const float*)d_in[0];
  const float* norm1_g  = (const float*)d_in[1];
  const float* norm1_b  = (const float*)d_in[2];
  const float* qkv_w    = (const float*)d_in[3];
  const float* qkv_b    = (const float*)d_in[4];
  const float* rel_tab  = (const float*)d_in[5];
  const float* proj_w   = (const float*)d_in[6];
  const float* proj_b   = (const float*)d_in[7];
  const float* norm2_g  = (const float*)d_in[8];
  const float* norm2_b  = (const float*)d_in[9];
  const float* fc1_w    = (const float*)d_in[10];
  const float* fc1_b    = (const float*)d_in[11];
  const float* fc2_w    = (const float*)d_in[12];
  const float* fc2_b    = (const float*)d_in[13];
  const int*   rel_idx  = (const int*)d_in[14];
  float* out = (float*)d_out;

  char* ws = (char*)d_ws;
  // region 0: converted weights + expanded bias (< 2 MB)
  __bf16* wq   = (__bf16*)(ws + 0);                    // 768*256
  __bf16* wp   = (__bf16*)(ws + 393216);               // 256*256
  __bf16* w1   = (__bf16*)(ws + 393216 + 131072);      // 1024*256
  __bf16* w2   = (__bf16*)(ws + 393216 + 131072 + 524288);
  float*  bias = (float*)(ws + 393216 + 131072 + 524288 + 524288);  // 8*64*64
  const size_t offA = 2ull << 20;                      // 128 MB region
  __bf16* qkv      = (__bf16*)(ws + offA);                         // 65536*768 bf16 (96 MB)
  __bf16* attn_out = (__bf16*)(ws + offA + 100663296ull);          // 65536*256 bf16 (32 MB)
  __bf16* fc1_out  = (__bf16*)(ws + offA);                         // aliases qkv+attn (dead)
  const size_t offB = offA + 134217728ull;
  __bf16* y_win    = (__bf16*)(ws + offB);                         // 32 MB
  __bf16* ln2_out  = (__bf16*)(ws + offB);                         // aliases y_win (dead)
  const size_t offC = offB + 33554432ull;
  float*  xs2      = (float*)(ws + offC);                          // 64 MB

  swin_prep<<<1024, 256, 0, stream>>>(qkv_w, proj_w, fc1_w, fc2_w, rel_tab, rel_idx,
                                      wq, wp, w1, w2, bias);
  swin_ln1<<<8192, 256, 0, stream>>>(x, norm1_g, norm1_b, y_win);
  swin_qkv<<<6144, 256, 0, stream>>>(y_win, wq, qkv_b, qkv);         // 512x12 tiles
  swin_attn<<<8192, 128, 0, stream>>>(qkv, bias, attn_out);
  swin_proj<<<2048, 256, 0, stream>>>(attn_out, wp, proj_b, x, xs2); // 512x4 tiles
  swin_ln2<<<8192, 256, 0, stream>>>(xs2, norm2_g, norm2_b, ln2_out);
  swin_fc1<<<8192, 256, 0, stream>>>(ln2_out, w1, fc1_b, fc1_out);   // 512x16 tiles
  swin_fc2<<<2048, 256, 0, stream>>>(fc1_out, w2, fc2_b, xs2, out);  // 512x4 tiles
}
